// SynthesisConv_29308856828622
// MI455X (gfx1250) — matrix operand order repaired and checked by reading, not compiled, not run on hardware
//
#include <hip/hip_runtime.h>
#include <math.h>

// ---------------------------------------------------------------------------
// StyleGAN2 SynthesisConv (modulated conv_transpose x2 + FIR + noise/bias/
// lrelu/clamp) for MI455X (gfx1250, wave32, WMMA + async-to-LDS).
//
//   style[b,cin]  = w_latent @ (affine_w/sqrt(512)).T + affine_b
//   dcoef[b,cout] = rsqrt( sum_cin style^2 * Wsq[cout,cin] + 1e-8 )
//   y65[b,co]     = dcoef[b,co] * conv_transpose( x * style[b,cin], conv_w )
//   out           = clamp(lrelu(FIR4x4(y65) + noise*ns + bias) * sqrt2, 256)
//
// conv_transpose decomposes into 4 parity phases (py,px), tap extents
// KHxKW = (2-py)x(2-px).  GEMM K index is TAP-MAJOR: k = tap*512 + cin, so
// every 32-wide K chunk has a fixed tap and 32 consecutive cin.  With
// channels-last xs2[b][h][w][cin] (f16) and phase-packed weights
// apack[phase][cout][k] (f16), both LDS tiles stage as aligned b128 segments,
// moved with GLOBAL_LOAD_ASYNC_TO_LDS_B128 into double-buffered LDS
// (ASYNCcnt pipeline), and consumed by v_wmma_f32_16x16x32_f16.
// ---------------------------------------------------------------------------

typedef _Float16 v8h  __attribute__((ext_vector_type(8)));
typedef _Float16 v16h __attribute__((ext_vector_type(16)));
typedef float    v8f  __attribute__((ext_vector_type(8)));

#define BATCH 16
#define CIN   512
#define COUT  512
#define HH    32
#define WDIM  512

// apack phase offsets in halves: phase sizes 512*{2048,1024,1024,512}
#define APH0 0
#define APH1 1048576
#define APH2 1572864
#define APH3 2097152
#define APACK_HALVES 2359296   // 4.5 MB

// ---------------------------------------------------------------- style -----
__global__ void style_kernel(const float* __restrict__ wlat,
                             const float* __restrict__ aw,
                             const float* __restrict__ ab,
                             float* __restrict__ style) {
  int idx = blockIdx.x * blockDim.x + threadIdx.x;   // [0, 16*512)
  int b = idx >> 9, c = idx & 511;
  const float* wl = wlat + b * WDIM;
  const float* aww = aw + c * WDIM;
  float acc = 0.f;
#pragma unroll 4
  for (int k = 0; k < WDIM; ++k) acc += wl[k] * aww[k];
  style[idx] = acc * 0.044194173824159216f /* 1/sqrt(512) */ + ab[c];
}

// ------------------------- Wsq + phase-packed f16 weights -------------------
__global__ void wsq_pack_kernel(const float* __restrict__ w,
                                float* __restrict__ wsq,
                                _Float16* __restrict__ apack) {
  int idx = blockIdx.x * blockDim.x + threadIdx.x;   // [0, 512*512)
  int co = idx >> 9, cin = idx & 511;
  const float* p = w + idx * 9;                      // w[co][cin][3][3]
  float w9[9];
  float s = 0.f;
#pragma unroll
  for (int i = 0; i < 9; ++i) {
    w9[i] = p[i];
    s += w9[i] * w9[i];
  }
  wsq[idx] = s;
  // phase (0,0): taps (a,bb) 2x2, k = (a*2+bb)*512+cin, val w[2-2a][2-2bb]
#pragma unroll
  for (int a = 0; a < 2; ++a)
#pragma unroll
    for (int bb = 0; bb < 2; ++bb)
      apack[APH0 + (size_t)co * 2048 + (a * 2 + bb) * 512 + cin] =
          (_Float16)w9[(2 - 2 * a) * 3 + (2 - 2 * bb)];
  // phase (0,1): taps a, val w[2-2a][1]
#pragma unroll
  for (int a = 0; a < 2; ++a)
    apack[APH1 + (size_t)co * 1024 + a * 512 + cin] =
        (_Float16)w9[(2 - 2 * a) * 3 + 1];
  // phase (1,0): taps bb, val w[1][2-2bb]
#pragma unroll
  for (int bb = 0; bb < 2; ++bb)
    apack[APH2 + (size_t)co * 1024 + bb * 512 + cin] =
        (_Float16)w9[3 + (2 - 2 * bb)];
  // phase (1,1): w[1][1]
  apack[APH3 + (size_t)co * 512 + cin] = (_Float16)w9[4];
}

// ---------------------------------------------------------------- dcoef -----
__global__ void dcoef_kernel(const float* __restrict__ style,
                             const float* __restrict__ wsq,
                             float* __restrict__ dcoef) {
  int idx = blockIdx.x * blockDim.x + threadIdx.x;   // [0, 16*512)
  int b = idx >> 9, co = idx & 511;
  float s = 1e-8f;
  const float* st = style + b * CIN;
  const float* wq = wsq + co * CIN;
#pragma unroll 4
  for (int cin = 0; cin < CIN; ++cin) {
    float t = st[cin];
    s += t * t * wq[cin];
  }
  dcoef[idx] = rsqrtf(s);
}

// ----------------------- x * style -> f16, channels-last --------------------
__global__ void xscale_kernel(const float* __restrict__ x,
                              const float* __restrict__ style,
                              _Float16* __restrict__ xs2) {
  int idx = blockIdx.x * blockDim.x + threadIdx.x;   // [0, 16*1024*512)
  int cin = idx & 511;
  int pix = idx >> 9;                                // b*1024 + h*32 + w
  int b = pix >> 10;
  int hw = pix & 1023;
  xs2[idx] = (_Float16)(x[(((b << 9) + cin) << 10) + hw] *
                        style[(b << 9) + cin]);
}

// ------------------------------------------------ zero pad block (256 B) ----
__global__ void zeropad_kernel(_Float16* __restrict__ zp) {
  zp[threadIdx.x] = (_Float16)0.f;                   // <<<1,128>>>
}

// ---------------------------------------------- WMMA phase GEMM kernel ------
// grid: x = 18 pixel tiles (64 px), y = 4 cout tiles (128 ch), z = b*4+phase
// block: 256 threads = 8 waves (4 Mwaves x 2 Nwaves); wave tile 32x32.
// Double-buffered LDS filled by global_load_async_to_lds_b128 (ASYNCcnt),
// 4 x v_wmma_f32_16x16x32_f16 per wave per K chunk.
__global__ __launch_bounds__(256)
void upconv_wmma_kernel(const _Float16* __restrict__ xs2,
                        const _Float16* __restrict__ apack,
                        const float* __restrict__ dcoef,
                        const _Float16* __restrict__ zeropad,
                        float* __restrict__ y65) {
  const int bp = blockIdx.z;
  const int b  = bp >> 2;
  const int py = (bp >> 1) & 1;
  const int px = bp & 1;
  const int kwLog = px ? 0 : 1;
  const int KW = 1 << kwLog;
  const int kkLog = (py ? 0 : 1) + kwLog;
  const int U = py ? 32 : 33;
  const int V = px ? 32 : 33;
  const int NP = U * V;
  const int coutBase = blockIdx.y * 128;
  const int nBase = blockIdx.x * 64;
  if (nBase >= NP) return;               // uniform over block: safe early exit
  const int Ktot = CIN << kkLog;         // {2048,1024,1024,512}

  const size_t phOffTab[4] = {APH0, APH1, APH2, APH3};
  const _Float16* Aph = apack + phOffTab[(py << 1) | px];

  __shared__ _Float16 Ash[2][128][40];   // [buf][cout][k] (+8 halves pad)
  __shared__ _Float16 Bsh[2][64][40];    // [buf][pixel][k]

  const int tid = threadIdx.x;
  const int lane = tid & 31;
  const int laneLo = lane & 15;
  const int laneHi = lane >> 4;
  const int wave = tid >> 5;
  const int wm = wave >> 1;              // 0..3 -> M rows [wm*32, +32)
  const int wn = wave & 1;               // 0..1 -> N cols [wn*32, +32)

  // ---- fixed staging coordinates (one b128 segment per thread per tile) ----
  const int am0 = tid >> 2;              // A rows: am0 and am0+64
  const int aseg = (tid & 3) << 3;       // half offset within 32-k row
  const _Float16* aPtr0 = Aph + (size_t)(coutBase + am0) * Ktot + aseg;
  const _Float16* aPtr1 = aPtr0 + (size_t)64 * Ktot;

  const int bn = tid >> 2;               // B pixel row 0..63
  const int bng = nBase + bn;
  const bool bvalid = bng < NP;
  int bu = 0, bv = 0;
  if (bvalid) { bu = bng / V; bv = bng - bu * V; }
  const int oy0 = py ? 0 : 1, ox0 = px ? 0 : 1;
  const _Float16* xsBase = xs2 + (size_t)b * (HH * HH * CIN) + aseg;
  const _Float16* zpad = zeropad + aseg;

  // per-thread LDS byte addresses (generic LDS pointer truncates to LDS addr)
  unsigned ldsA0[2], ldsA1[2], ldsB[2];
  ldsA0[0] = (unsigned)(size_t)&Ash[0][am0][aseg];
  ldsA0[1] = (unsigned)(size_t)&Ash[1][am0][aseg];
  ldsA1[0] = (unsigned)(size_t)&Ash[0][am0 + 64][aseg];
  ldsA1[1] = (unsigned)(size_t)&Ash[1][am0 + 64][aseg];
  ldsB[0]  = (unsigned)(size_t)&Bsh[0][bn][aseg];
  ldsB[1]  = (unsigned)(size_t)&Bsh[1][bn][aseg];

// issue chunk K0 into LDS buffer BUF: 3 async b128 per thread, EXEC uniform
// (OOB im2col lanes read from the zeroed pad block instead of masking).
#define ISSUE(K0, BUF)                                                        \
  do {                                                                        \
    const int tap_ = (K0) >> 9;                                               \
    const int cin0_ = (K0) & 511;                                             \
    const _Float16* gA0_ = aPtr0 + (K0);                                      \
    const _Float16* gA1_ = aPtr1 + (K0);                                      \
    const int a_ = tap_ >> kwLog;                                             \
    const int bb_ = tap_ & (KW - 1);                                          \
    const int xr_ = bu + a_ - oy0;                                            \
    const int xc_ = bv + bb_ - ox0;                                           \
    const _Float16* gB_ =                                                     \
        (bvalid && (unsigned)xr_ < 32u && (unsigned)xc_ < 32u)                \
            ? (xsBase + ((((xr_ << 5) + xc_) << 9) + cin0_))                  \
            : zpad;                                                           \
    asm volatile("global_load_async_to_lds_b128 %0, %1, off"                  \
                 ::"v"(ldsA0[BUF]), "v"(gA0_) : "memory");                    \
    asm volatile("global_load_async_to_lds_b128 %0, %1, off"                  \
                 ::"v"(ldsA1[BUF]), "v"(gA1_) : "memory");                    \
    asm volatile("global_load_async_to_lds_b128 %0, %1, off"                  \
                 ::"v"(ldsB[BUF]), "v"(gB_) : "memory");                      \
  } while (0)

#define WAIT_ASYNC() asm volatile("s_wait_asynccnt 0x0" ::: "memory")

// fragments per ISA 7.12.2 (A: lanes<16 K=0..7,16..23, lanes>=16 +8;
// B: lane = N, same K order as A), then 4 WMMAs.
#define COMPUTE(BUF)                                                          \
  do {                                                                        \
    const _Float16* ar0_ = &Ash[BUF][wm * 32 + laneLo][0];                    \
    const _Float16* ar1_ = &Ash[BUF][wm * 32 + 16 + laneLo][0];               \
    v16h af0_ = __builtin_shufflevector(                                      \
        *(const v8h*)(ar0_ + laneHi * 8),                                     \
        *(const v8h*)(ar0_ + 16 + laneHi * 8),                                \
        0, 1, 2, 3, 4, 5, 6, 7, 8, 9, 10, 11, 12, 13, 14, 15);                \
    v16h af1_ = __builtin_shufflevector(                                      \
        *(const v8h*)(ar1_ + laneHi * 8),                                     \
        *(const v8h*)(ar1_ + 16 + laneHi * 8),                                \
        0, 1, 2, 3, 4, 5, 6, 7, 8, 9, 10, 11, 12, 13, 14, 15);                \
    const _Float16* br0_ = &Bsh[BUF][wn * 32 + laneLo][0];                    \
    const _Float16* br1_ = &Bsh[BUF][wn * 32 + 16 + laneLo][0];               \
    v16h bf0_ = __builtin_shufflevector(                                      \
        *(const v8h*)(br0_ + laneHi * 8),                                     \
        *(const v8h*)(br0_ + 16 + laneHi * 8),                                \
        0, 1, 2, 3, 4, 5, 6, 7, 8, 9, 10, 11, 12, 13, 14, 15);                \
    v16h bf1_ = __builtin_shufflevector(                                      \
        *(const v8h*)(br1_ + laneHi * 8),                                     \
        *(const v8h*)(br1_ + 16 + laneHi * 8),                                \
        0, 1, 2, 3, 4, 5, 6, 7, 8, 9, 10, 11, 12, 13, 14, 15);                \
    acc0 = __builtin_amdgcn_wmma_f32_16x16x32_f16(false, af0_, false, bf0_,   \
                                                  (short)0, acc0, false,      \
                                                  false);                     \
    acc1 = __builtin_amdgcn_wmma_f32_16x16x32_f16(false, af0_, false, bf1_,   \
                                                  (short)0, acc1, false,      \
                                                  false);                     \
    acc2 = __builtin_amdgcn_wmma_f32_16x16x32_f16(false, af1_, false, bf0_,   \
                                                  (short)0, acc2, false,      \
                                                  false);                     \
    acc3 = __builtin_amdgcn_wmma_f32_16x16x32_f16(false, af1_, false, bf1_,   \
                                                  (short)0, acc3, false,      \
                                                  false);                     \
  } while (0)

  v8f acc0 = {}, acc1 = {}, acc2 = {}, acc3 = {};

  const int nChunks = Ktot >> 5;         // 16/32/64: always even
  ISSUE(0, 0);
  for (int i = 0; i < nChunks; i += 2) {
    WAIT_ASYNC();                        // buf0 of this pair is resident
    __syncthreads();                     // all waves: data visible, buf1 free
    ISSUE((i + 1) << 5, 1);              // overlap with compute of buf0
    COMPUTE(0);
    WAIT_ASYNC();                        // buf1 resident
    __syncthreads();                     // all waves done reading buf0
    if (i + 2 < nChunks) ISSUE((i + 2) << 5, 0);
    COMPUTE(1);
  }
#undef ISSUE
#undef WAIT_ASYNC
#undef COMPUTE

  // ---- epilogue: demodulate, scatter to parity positions of y65 ----
  const int ngA = nBase + wn * 32 + laneLo;        // fn = 0
  const int ngB = ngA + 16;                        // fn = 1
  int uA = 0, vA = 0, uB = 0, vB = 0;
  const bool okA = ngA < NP, okB = ngB < NP;
  if (okA) { uA = ngA / V; vA = ngA - uA * V; }
  if (okB) { uB = ngB / V; vB = ngB - uB * V; }
  const float* dc = dcoef + b * COUT;
  float* ybase = y65 + (size_t)(b * COUT) * (65 * 65);

#pragma unroll
  for (int r = 0; r < 8; ++r) {
    int co0 = coutBase + wm * 32 + 8 * laneHi + r;       // fm = 0 frag rows
    int co1 = co0 + 16;                                  // fm = 1 frag rows
    float d0 = dc[co0], d1 = dc[co1];
    if (okA) {
      size_t off = (size_t)(2 * uA + py) * 65 + (2 * vA + px);
      ybase[(size_t)co0 * 4225 + off] = acc0[r] * d0;
      ybase[(size_t)co1 * 4225 + off] = acc2[r] * d1;
    }
    if (okB) {
      size_t off = (size_t)(2 * uB + py) * 65 + (2 * vB + px);
      ybase[(size_t)co0 * 4225 + off] = acc1[r] * d0;
      ybase[(size_t)co1 * 4225 + off] = acc3[r] * d1;
    }
  }
}

// --------------------------- FIR 4x4 + noise + bias + lrelu + clamp ---------
__global__ void fir_epilogue_kernel(const float* __restrict__ y65,
                                    const float* __restrict__ noise,
                                    const float* __restrict__ nstr,
                                    const float* __restrict__ bias,
                                    float* __restrict__ out) {
  int idx = blockIdx.x * blockDim.x + threadIdx.x;
  if (idx >= BATCH * COUT * 64 * 64) return;
  int c  = idx & 63;
  int r  = (idx >> 6) & 63;
  int co = (idx >> 12) & 511;
  int b  = idx >> 21;
  const float* base = y65 + (size_t)(b * COUT + co) * 4225;
  const float fco[4] = {1.f, 3.f, 3.f, 1.f};   // f2 = outer(f)/16
  float acc = 0.f;
#pragma unroll
  for (int i = 0; i < 4; ++i) {
    int rr = r - 1 + i;
    if ((unsigned)rr >= 65u) continue;
    float rs = 0.f;
#pragma unroll
    for (int j = 0; j < 4; ++j) {
      int cc = c - 1 + j;
      if ((unsigned)cc >= 65u) continue;
      rs += fco[j] * base[rr * 65 + cc];
    }
    acc += fco[i] * rs;
  }
  acc *= (1.f / 16.f);
  acc += noise[(b * 64 + r) * 64 + c] * nstr[0];
  acc += bias[co];
  float y = (acc > 0.f ? acc : 0.2f * acc) * 1.4142135623730951f;
  out[idx] = fminf(fmaxf(y, -256.f), 256.f);
}

// ---------------------------------------------------------------------------
extern "C" void kernel_launch(void* const* d_in, const int* in_sizes, int n_in,
                              void* d_out, int out_size, void* d_ws,
                              size_t ws_size, hipStream_t stream) {
  (void)in_sizes; (void)n_in; (void)out_size; (void)ws_size;
  const float* x     = (const float*)d_in[0];
  const float* wlat  = (const float*)d_in[1];
  const float* aw    = (const float*)d_in[2];
  const float* ab    = (const float*)d_in[3];
  const float* cw    = (const float*)d_in[4];
  const float* noise = (const float*)d_in[5];
  const float* nstr  = (const float*)d_in[6];
  const float* bias  = (const float*)d_in[7];
  float* out = (float*)d_out;

  // workspace layout (256-B aligned chunks)
  char* ws = (char*)d_ws;
  float* style = (float*)ws;       ws += (size_t)BATCH * CIN * 4;           // 32 KB
  float* dcoef = (float*)ws;       ws += (size_t)BATCH * COUT * 4;          // 32 KB
  float* wsq   = (float*)ws;       ws += (size_t)COUT * CIN * 4;            // 1 MB
  _Float16* apack = (_Float16*)ws; ws += (size_t)APACK_HALVES * 2;          // 4.5 MB
  _Float16* xs2   = (_Float16*)ws; ws += (size_t)BATCH * HH * HH * CIN * 2; // 16 MB
  _Float16* zpad  = (_Float16*)ws; ws += 256;                               // 256 B
  float* y65 = (float*)ws;         // 16*512*65*65*4 = 138.4 MB

  style_kernel<<<BATCH * CIN / 256, 256, 0, stream>>>(wlat, aw, ab, style);
  wsq_pack_kernel<<<COUT * CIN / 256, 256, 0, stream>>>(cw, wsq, apack);
  dcoef_kernel<<<BATCH * COUT / 256, 256, 0, stream>>>(style, wsq, dcoef);
  xscale_kernel<<<BATCH * HH * HH * CIN / 256, 256, 0, stream>>>(x, style, xs2);
  zeropad_kernel<<<1, 128, 0, stream>>>(zpad);

  dim3 gconv(18, COUT / 128, BATCH * 4);  // pixel tiles x cout tiles x (b,phase)
  upconv_wmma_kernel<<<gconv, 256, 0, stream>>>(xs2, apack, dcoef, zpad, y65);

  int nOut = BATCH * COUT * 64 * 64;
  fir_epilogue_kernel<<<(nOut + 255) / 256, 256, 0, stream>>>(y65, noise, nstr,
                                                              bias, out);
}
